// OurConv3d_34041910788791
// MI455X (gfx1250) — compile-verified, matrix operand order checked by reading, not run on hardware
//
#include <hip/hip_runtime.h>

// ---------------------------------------------------------------------------
// OurConv3d (quadratic B-spline splat-gather conv) for MI455X / gfx1250.
//
// Pass 1: transpose + downconvert grid [B,C=32,96^3] f32 -> [B,96^3,C] f16
//         into d_ws (113.2 MB).
// Pass 2: build WMMA-B fragment-swizzled f16 kernel matrix (55 KB) in d_ws.
// Pass 3: per wave32, one 32-point tile (2 M-tiles sharing B fragments):
//         A[32,864] (f16, spline weights folded in, built in-register from
//         global b128 gathers) x B[864,32] (f16, LDS, fragment layout,
//         conflict-free) via 27 taps x 4 v_wmma_f32_16x16x32_f16,
//         + sum(W27)*bias epilogue.
// ---------------------------------------------------------------------------

typedef __attribute__((ext_vector_type(16))) _Float16 v16h;
typedef __attribute__((ext_vector_type(8)))  _Float16 v8h;
typedef __attribute__((ext_vector_type(8)))  float    v8f;
typedef __attribute__((ext_vector_type(4)))  float    v4f;

#define GDIM   96
#define GS     (96 * 96 * 96)      // 884736 cells per batch
#define CIN    32
#define COUT   32
#define NTAPS  27
#define NBATCH 2
#define BSWZ_N (NTAPS * 2 * 2 * 32 * 8)   // 27648 f16 = 55296 B

// ---------------------------------------------------------------------------
// Kernel 1: channels-first f32 -> channels-last f16  (coalesced via LDS tile)
// grid: (GS/64, B), block: 256
// ---------------------------------------------------------------------------
__global__ __launch_bounds__(256) void grid_transpose_f16(
    const float* __restrict__ in, _Float16* __restrict__ out)
{
    __shared__ float tile[CIN][65];            // +1 pad: kill bank conflicts
    const int b  = blockIdx.y;
    const int s0 = blockIdx.x * 64;
    const int tid = threadIdx.x;

    const float* inb = in + (size_t)b * CIN * GS;

    const int l = tid & 63;
    const int g = tid >> 6;                    // 0..3
#pragma unroll
    for (int c = g; c < CIN; c += 4)
        tile[c][l] = inb[(size_t)c * GS + s0 + l];

    __syncthreads();

    const int sl = tid >> 2;                   // 0..63
    const int cg = (tid & 3) * 8;              // 0,8,16,24
    v8h v;
#pragma unroll
    for (int k = 0; k < 8; ++k)
        v[k] = (_Float16)tile[cg + k][sl];
    *(v8h*)(out + ((size_t)b * GS + s0 + sl) * CIN + cg) = v;
}

// ---------------------------------------------------------------------------
// Kernel 2: one-off swizzle of the conv kernel into WMMA-B fragment layout.
// Layout: bsw[tap][tile][half][lane][8]  (half = i>>3 of the v16h fragment)
//   fragment element i of lane L encodes K = hi*8 + (i&7) + (i>=8 ? 16 : 0),
//   N = tile*16 + (L&15), hi = L>>4   (ISA 16-bit B-matrix striping).
// ---------------------------------------------------------------------------
__global__ __launch_bounds__(256) void build_bswz(
    const float* __restrict__ kern, _Float16* __restrict__ bsw)
{
    const int tap = blockIdx.x;
    const int tid = threadIdx.x;
    const int src = tid * 4;                          // within tap: 0..1023
    const v4f v   = *(const v4f*)(kern + tap * (CIN * COUT) + src);
    const int kin = src >> 5;                         // input channel
    const int c0  = src & 31;                         // output channel base
    const int hi  = (kin >> 3) & 1;
    const int i   = (kin & 7) | ((kin & 16) >> 1);    // fragment element
    const int half = i >> 3;
#pragma unroll
    for (int c = 0; c < 4; ++c) {
        const int cout = c0 + c;
        const int tile = cout >> 4;
        const int lane = hi * 16 + (cout & 15);
        bsw[(((tap * 2 + tile) * 2 + half) * 32 + lane) * 8 + (i & 7)] =
            (_Float16)v[c];
    }
}

// ---------------------------------------------------------------------------
// Kernel 3: main splat-gather GEMM.
// block = 256 threads = 8 wave32; each wave handles 32 points of one batch.
// grid: (ceil(wavesPerB/8), NBATCH)
// ---------------------------------------------------------------------------
__global__ __launch_bounds__(256) void splat_conv_wmma(
    const _Float16* __restrict__ gridh,   // [B, GS, 32] f16 (pass 1)
    const _Float16* __restrict__ bsw,     // swizzled B (pass 2)
    const float*    __restrict__ pos,     // [B, N, 3]
    const float*    __restrict__ bias,    // [32]
    float*          __restrict__ out,     // [B, N, 32]
    int Npts, int wavesPerB)
{
    __shared__ __align__(16) _Float16 bsh[BSWZ_N];   // 55296 B
    __shared__ float sumw[8][32];

    const int tid = threadIdx.x;

    // ---- LDS fill: fully unrolled b128 copies (independent loads) ---------
    {
        const int e0 = tid * 8;
#pragma unroll
        for (int k = 0; k < 13; ++k)
            *(v8h*)(bsh + e0 + k * 2048) = *(const v8h*)(bsw + e0 + k * 2048);
        if (tid < 128)
            *(v8h*)(bsh + e0 + 13 * 2048) = *(const v8h*)(bsw + e0 + 13 * 2048);
    }
    __syncthreads();

    const int wid  = tid >> 5;
    const int lane = tid & 31;
    const int gw   = blockIdx.x * 8 + wid;
    if (gw >= wavesPerB) return;

    const int b  = blockIdx.y;
    const int p0 = gw * 32;

    const int m  = lane & 15;        // point-in-tile for this lane
    const int hi = lane >> 4;        // which K half-set this lane provides

    // ---- per-point quadratic B-spline weights (two points per lane) -------
    float wq[2][3][3];
    const _Float16* abase[2];
    const _Float16* gb = gridh + (size_t)b * (GS * CIN);
#pragma unroll
    for (int pt = 0; pt < 2; ++pt) {
        const int p = min(p0 + pt * 16 + m, Npts - 1);
        const float* pp = pos + ((size_t)b * Npts + p) * 3;
        int ib[3];
#pragma unroll
        for (int ax = 0; ax < 3; ++ax) {
            const float Xp = pp[ax] / 0.1f;      // (pos - GRID_MIN) / DX
            const float bf = floorf(Xp);
            ib[ax] = (int)bf;
            const float fx = Xp - bf - 0.5f;
            wq[pt][ax][0] = 0.5f * (0.5f - fx) * (0.5f - fx);
            wq[pt][ax][1] = 0.75f - fx * fx;
            wq[pt][ax][2] = 0.5f * (0.5f + fx) * (0.5f + fx);
        }
        sumw[wid][pt * 16 + m] =
            (wq[pt][0][0] + wq[pt][0][1] + wq[pt][0][2]) *
            (wq[pt][1][0] + wq[pt][1][1] + wq[pt][1][2]) *
            (wq[pt][2][0] + wq[pt][2][1] + wq[pt][2][2]);
        const int fl = ((ib[0] - 1) * GDIM + (ib[1] - 1)) * GDIM + (ib[2] - 1);
        abase[pt] = gb + (size_t)fl * CIN + hi * 8;
    }

    v8f c00 = {}, c01 = {};          // M-tile 0 x N-tiles {0,1}
    v8f c10 = {}, c11 = {};          // M-tile 1 x N-tiles {0,1}

#pragma unroll
    for (int i = 0; i < 3; ++i) {
#pragma unroll
        for (int j = 0; j < 3; ++j) {
#pragma unroll
            for (int l = 0; l < 3; ++l) {
                const int tap = (i * 3 + j) * 3 + l;
                const int off = (i * (GDIM * GDIM) + j * GDIM + l) * CIN;

                // shared B fragments (conflict-free 16 B/lane LDS reads)
                const v8h b0a = *(const v8h*)(bsh + (((tap*2+0)*2+0)*32 + lane)*8);
                const v8h b0b = *(const v8h*)(bsh + (((tap*2+0)*2+1)*32 + lane)*8);
                const v8h b1a = *(const v8h*)(bsh + (((tap*2+1)*2+0)*32 + lane)*8);
                const v8h b1b = *(const v8h*)(bsh + (((tap*2+1)*2+1)*32 + lane)*8);
                const v16h b0 = __builtin_shufflevector(
                    b0a, b0b, 0,1,2,3,4,5,6,7,8,9,10,11,12,13,14,15);
                const v16h b1 = __builtin_shufflevector(
                    b1a, b1b, 0,1,2,3,4,5,6,7,8,9,10,11,12,13,14,15);

                // M-tile 0
                {
                    const _Float16 wh =
                        (_Float16)(wq[0][0][i] * wq[0][1][j] * wq[0][2][l]);
                    const _Float16* r = abase[0] + off;
                    const v8h lo = *(const v8h*)r;
                    const v8h hh = *(const v8h*)(r + 16);
                    const v8h plo = lo * wh;             // v_pk_mul_f16
                    const v8h phi = hh * wh;
                    const v16h a = __builtin_shufflevector(
                        plo, phi, 0,1,2,3,4,5,6,7,8,9,10,11,12,13,14,15);
                    c00 = __builtin_amdgcn_wmma_f32_16x16x32_f16(
                            false, a, false, b0, (short)0, c00, false, false);
                    c01 = __builtin_amdgcn_wmma_f32_16x16x32_f16(
                            false, a, false, b1, (short)0, c01, false, false);
                }
                // M-tile 1
                {
                    const _Float16 wh =
                        (_Float16)(wq[1][0][i] * wq[1][1][j] * wq[1][2][l]);
                    const _Float16* r = abase[1] + off;
                    const v8h lo = *(const v8h*)r;
                    const v8h hh = *(const v8h*)(r + 16);
                    const v8h plo = lo * wh;
                    const v8h phi = hh * wh;
                    const v16h a = __builtin_shufflevector(
                        plo, phi, 0,1,2,3,4,5,6,7,8,9,10,11,12,13,14,15);
                    c10 = __builtin_amdgcn_wmma_f32_16x16x32_f16(
                            false, a, false, b0, (short)0, c10, false, false);
                    c11 = __builtin_amdgcn_wmma_f32_16x16x32_f16(
                            false, a, false, b1, (short)0, c11, false, false);
                }
            }
        }
    }

    // ---- epilogue: D layout is M = r + 8*hi per VGPR, N = lane&15 ---------
    const int n = lane & 15;
    const float bias0 = bias[n];
    const float bias1 = bias[16 + n];
    const size_t obase = (size_t)b * Npts;
#pragma unroll
    for (int r = 0; r < 8; ++r) {
        const int M = hi * 8 + r;
        {   // M-tile 0
            const int pr = p0 + M;
            if (pr < Npts) {
                const float sw = sumw[wid][M];
                const size_t o = (obase + pr) * COUT;
                out[o + n]      = c00[r] + sw * bias0;
                out[o + 16 + n] = c01[r] + sw * bias1;
            }
        }
        {   // M-tile 1
            const int pr = p0 + 16 + M;
            if (pr < Npts) {
                const float sw = sumw[wid][16 + M];
                const size_t o = (obase + pr) * COUT;
                out[o + n]      = c10[r] + sw * bias0;
                out[o + 16 + n] = c11[r] + sw * bias1;
            }
        }
    }
}

// ---------------------------------------------------------------------------
extern "C" void kernel_launch(void* const* d_in, const int* in_sizes, int n_in,
                              void* d_out, int out_size, void* d_ws, size_t ws_size,
                              hipStream_t stream)
{
    (void)n_in; (void)out_size; (void)ws_size;
    const float* inp  = (const float*)d_in[0];   // [2, 32, 96,96,96]
    const float* pos  = (const float*)d_in[1];   // [2, N, 3]
    const float* kern = (const float*)d_in[2];   // [3,3,3,32,32]
    const float* bias = (const float*)d_in[3];   // [32]

    const int Npts = in_sizes[1] / (NBATCH * 3); // 100000

    // workspace: [grid f16: 113,246,208 B][bswz f16: 55,296 B]
    _Float16* gridh = (_Float16*)d_ws;
    _Float16* bsw   = gridh + (size_t)NBATCH * GS * CIN;

    // Pass 1: transpose + f32->f16 downconvert (channels-last)
    dim3 g1(GS / 64, NBATCH);
    grid_transpose_f16<<<g1, 256, 0, stream>>>(inp, gridh);

    // Pass 2: one-off B-fragment swizzle (27 blocks, trivial)
    build_bswz<<<NTAPS, 256, 0, stream>>>(kern, bsw);

    // Pass 3: WMMA splat-gather conv, 32 points per wave
    const int wavesPerB = (Npts + 31) / 32;      // 3125
    dim3 g3((wavesPerB + 7) / 8, NBATCH);
    splat_conv_wmma<<<g3, 256, 0, stream>>>(
        gridh, bsw, pos, bias, (float*)d_out, Npts, wavesPerB);
}
